// ScoringFunction_62663572849130
// MI455X (gfx1250) — matrix operand order repaired and checked by reading, not compiled, not run on hardware
//
#include <hip/hip_runtime.h>
#include <hip/hip_bf16.h>

typedef __attribute__((ext_vector_type(16))) _Float16 v16h;
typedef __attribute__((ext_vector_type(8)))  float    v8f;
typedef __attribute__((ext_vector_type(4)))  float    vf4;
typedef __attribute__((ext_vector_type(2)))  __fp16   v2fp16;

#define DIM     128            // embedding dim D
#define DLEN    256            // doc tokens per doc
#define QTOK    32             // query tokens
#define NTILES  (DLEN / 16)    // 16 column tiles of doc tokens
#define KCHUNKS (DIM / 32)     // 4 K-chunks of 32
#define WPB     8              // waves (docs) per block
#define NEGINF  (-__builtin_inff())

union V16 { v16h v; _Float16 h[16]; };

// Convert 8 consecutive f32 at p -> 8 f16 at dst (two b128 loads, packed cvt).
__device__ __forceinline__ void cvt8(const float* __restrict__ p, _Float16* dst) {
  const vf4* q = (const vf4*)p;
  vf4 f0 = q[0], f1 = q[1];
  v2fp16 a = __builtin_amdgcn_cvt_pkrtz(f0.x, f0.y);
  v2fp16 b = __builtin_amdgcn_cvt_pkrtz(f0.z, f0.w);
  v2fp16 c = __builtin_amdgcn_cvt_pkrtz(f1.x, f1.y);
  v2fp16 d = __builtin_amdgcn_cvt_pkrtz(f1.z, f1.w);
  dst[0] = (_Float16)a.x; dst[1] = (_Float16)a.y;
  dst[2] = (_Float16)b.x; dst[3] = (_Float16)b.y;
  dst[4] = (_Float16)c.x; dst[5] = (_Float16)c.y;
  dst[6] = (_Float16)d.x; dst[7] = (_Float16)d.y;
}

// Non-temporal variant for the stream-once doc embeddings (>L2 working set).
__device__ __forceinline__ void cvt8_nt(const float* __restrict__ p, _Float16* dst) {
  const vf4* q = (const vf4*)p;
  vf4 f0 = __builtin_nontemporal_load(q);
  vf4 f1 = __builtin_nontemporal_load(q + 1);
  v2fp16 a = __builtin_amdgcn_cvt_pkrtz(f0.x, f0.y);
  v2fp16 b = __builtin_amdgcn_cvt_pkrtz(f0.z, f0.w);
  v2fp16 c = __builtin_amdgcn_cvt_pkrtz(f1.x, f1.y);
  v2fp16 d = __builtin_amdgcn_cvt_pkrtz(f1.z, f1.w);
  dst[0] = (_Float16)a.x; dst[1] = (_Float16)a.y;
  dst[2] = (_Float16)b.x; dst[3] = (_Float16)b.y;
  dst[4] = (_Float16)c.x; dst[5] = (_Float16)c.y;
  dst[6] = (_Float16)d.x; dst[7] = (_Float16)d.y;
}

__global__ __launch_bounds__(WPB * 32)
void maxsim_wmma_kernel(const float* __restrict__ qe,
                        const unsigned char* __restrict__ qmask,
                        const float* __restrict__ de,
                        const unsigned char* __restrict__ dmask,
                        const int* __restrict__ ndocs_ptr,
                        float* __restrict__ out, int N) {
  const int wave = (int)(blockIdx.x * WPB + (threadIdx.x >> 5)); // doc index n
  const int lane = (int)(threadIdx.x & 31);
  if (wave >= N) return;                     // whole-wave uniform: EXEC stays full

  const int num_docs = *ndocs_ptr;           // scalar load, shared per wave
  const int b   = wave / num_docs;           // query block for this doc
  const int half = lane >> 4;                // 0: lanes 0-15, 1: lanes 16-31
  const int l15  = lane & 15;

  // ---- A fragments: query tiles, resident for whole doc -------------------
  // 16-bit A 16x32 layout: row = mt*16 + l15;
  // low lanes: K = kc*32 + {0..7, 16..23}; high lanes: K = kc*32 + {8..15, 24..31}
  V16 A[2][KCHUNKS];
  #pragma unroll
  for (int mt = 0; mt < 2; ++mt) {
    const float* qrow = qe + ((size_t)b * QTOK + (size_t)(mt * 16 + l15)) * DIM;
    #pragma unroll
    for (int kc = 0; kc < KCHUNKS; ++kc) {
      const int b1 = kc * 32 + (half ? 8  : 0);
      const int b2 = kc * 32 + (half ? 24 : 16);
      cvt8(qrow + b1, A[mt][kc].h);
      cvt8(qrow + b2, A[mt][kc].h + 8);
    }
  }

  // Running masked max per (row-register, column-lane)
  v8f rmax0, rmax1;
  #pragma unroll
  for (int r = 0; r < 8; ++r) { rmax0[r] = NEGINF; rmax1[r] = NEGINF; }

  const float* dbase = de + (size_t)wave * DLEN * DIM;
  const unsigned char* dm = dmask + (size_t)wave * DLEN;

  for (int nt = 0; nt < NTILES; ++nt) {
    const int col = nt * 16 + l15;                         // doc token (B column & C column)
    const float* drow = dbase + (size_t)col * DIM;         // Doc[col, :] contiguous

    // B fragments: 32x16 layout -> lane = column, low lanes K=kc*32+{0..7,16..23},
    // high lanes K=kc*32+{8..15,24..31} (matches A).
    V16 Bf[KCHUNKS];
    #pragma unroll
    for (int kc = 0; kc < KCHUNKS; ++kc) {
      const int kb = kc * 32 + (half ? 8 : 0);
      cvt8_nt(drow + kb,      Bf[kc].h);
      cvt8_nt(drow + kb + 16, Bf[kc].h + 8);
    }

    v8f c0 = {}, c1 = {};
    #pragma unroll
    for (int kc = 0; kc < KCHUNKS; ++kc) {
      c0 = __builtin_amdgcn_wmma_f32_16x16x32_f16(false, A[0][kc].v, false, Bf[kc].v,
                                                  (short)0, c0, false, false);
      c1 = __builtin_amdgcn_wmma_f32_16x16x32_f16(false, A[1][kc].v, false, Bf[kc].v,
                                                  (short)0, c1, false, false);
    }

    // Doc-token mask applies per column == per lane (same for all 8 row-regs).
    const bool ok = dm[col] != 0;
    #pragma unroll
    for (int r = 0; r < 8; ++r) {
      rmax0[r] = fmaxf(rmax0[r], ok ? c0[r] : NEGINF);
      rmax1[r] = fmaxf(rmax1[r], ok ? c1[r] : NEGINF);
    }
  }

  // Max over the 16 columns held by each 16-lane group (xor 1,2,4,8 stays in-group).
  #pragma unroll
  for (int off = 1; off < 16; off <<= 1) {
    #pragma unroll
    for (int r = 0; r < 8; ++r) {
      rmax0[r] = fmaxf(rmax0[r], __shfl_xor(rmax0[r], off, 32));
      rmax1[r] = fmaxf(rmax1[r], __shfl_xor(rmax1[r], off, 32));
    }
  }

  // C layout: VGPR r, lanes 0-15 -> row r; lanes 16-31 -> row r+8 (per M-tile).
  // Query-masked sum over the 32 query tokens.
  const unsigned char* qm = qmask + (size_t)b * QTOK;
  float s = 0.0f;
  #pragma unroll
  for (int r = 0; r < 8; ++r) {
    const int t0 = half * 8 + r;        // token from M-tile 0
    const int t1 = 16 + half * 8 + r;   // token from M-tile 1
    if (qm[t0]) s += rmax0[r];
    if (qm[t1]) s += rmax1[r];
  }
  s += __shfl_xor(s, 16, 32);           // combine the two half-wave partials
  if (lane == 0) out[wave] = s;
}

extern "C" void kernel_launch(void* const* d_in, const int* in_sizes, int n_in,
                              void* d_out, int out_size, void* d_ws, size_t ws_size,
                              hipStream_t stream) {
  (void)in_sizes; (void)n_in; (void)d_ws; (void)ws_size;
  const float*         qe = (const float*)d_in[0];
  const unsigned char* qm = (const unsigned char*)d_in[1];
  const float*         de = (const float*)d_in[2];
  const unsigned char* dm = (const unsigned char*)d_in[3];
  const int*           nd = (const int*)d_in[4];
  float* out = (float*)d_out;
  const int N = out_size;                       // number of docs (4096)
  const int blocks = (N + WPB - 1) / WPB;       // 1 wave32 per doc, 8 waves/block
  maxsim_wmma_kernel<<<blocks, WPB * 32, 0, stream>>>(qe, qm, de, dm, nd, out, N);
}
